// EncoderLayer_16604343567186
// MI455X (gfx1250) — matrix operand order repaired and checked by reading, not compiled, not run on hardware
//
#include <hip/hip_runtime.h>
#include <hip/hip_bf16.h>

// ---------------------------------------------------------------------------
// Transformer encoder layer (pre-LN, MHA + SiLU FFN) for gfx1250 / MI455X.
// fp32 in/out, f16 WMMA (v_wmma_f32_16x16x32_f16) for all matmuls.
// ---------------------------------------------------------------------------

typedef __attribute__((ext_vector_type(16))) _Float16 v16h;
typedef __attribute__((ext_vector_type(8)))  _Float16 v8h;
typedef __attribute__((ext_vector_type(8)))  float    v8f;

#define NB   4
#define SEQ  2048
#define DM   256
#define NH   8
#define DH   32
#define DFF  512
#define MROWS (NB * SEQ)      // 8192

// ---------------- WMMA helpers (CDNA5 16x16x32 f16 -> f32) -----------------

__device__ __forceinline__ v8f wmma_f32(v16h a, v16h b, v8f c) {
  // 8 args: (neg_a, A, neg_b, B, c_mod, C, reuse_a, reuse_b)
  return __builtin_amdgcn_wmma_f32_16x16x32_f16(false, a, false, b,
                                                (short)0, c, false, false);
}

// A-matrix fragment, 16x32 f16 (M x K), row-major source with leading dim ld.
// Lane layout (ISA 7.12.2): m = lane&15; lanes<16 hold K=[0..8)+[16..24),
// lanes>=16 hold K=[8..16)+[24..32).
__device__ __forceinline__ v16h load_a_frag(const _Float16* __restrict__ base,
                                            int ld, int lane) {
  const _Float16* p = base + (lane & 15) * ld + ((lane >> 4) << 3);
  v8h lo = *(const v8h*)p;
  v8h hi = *(const v8h*)(p + 16);
  return __builtin_shufflevector(lo, hi, 0, 1, 2, 3, 4, 5, 6, 7,
                                 8, 9, 10, 11, 12, 13, 14, 15);
}

// B-matrix fragment, 32x16 f16 (K x N). Source is W[N][K] row-major (so each
// lane n reads two K-runs of 8): lanes<16 -> K=[0..8)+[16..24),
// lanes>=16 -> K=[8..16)+[24..32).
__device__ __forceinline__ v16h load_b_frag(const _Float16* __restrict__ base,
                                            int ld, int lane) {
  const _Float16* p = base + (lane & 15) * ld + ((lane >> 4) << 3);
  v8h lo = *(const v8h*)p;
  v8h hi = *(const v8h*)(p + 16);
  return __builtin_shufflevector(lo, hi, 0, 1, 2, 3, 4, 5, 6, 7,
                                 8, 9, 10, 11, 12, 13, 14, 15);
}

// 8-wave GEMM core: block tile 128(M) x 64(N), wave tile 32x32 (2x2 frags).
__device__ __forceinline__ void gemm_core(const _Float16* __restrict__ A,
                                          const _Float16* __restrict__ W,
                                          int K, int row0, int col0, int lane,
                                          v8f (&acc)[2][2]) {
  const _Float16* a0p = A + (size_t)row0 * K;
  const _Float16* a1p = A + (size_t)(row0 + 16) * K;
  const _Float16* b0p = W + (size_t)col0 * K;
  const _Float16* b1p = W + (size_t)(col0 + 16) * K;
  for (int kt = 0; kt < K; kt += 32) {
    if (kt + 32 < K)  // next A tile -> global_prefetch_b8
      __builtin_prefetch(a0p + kt + 32, 0, 0);
    v16h a0 = load_a_frag(a0p + kt, K, lane);
    v16h a1 = load_a_frag(a1p + kt, K, lane);
    v16h b0 = load_b_frag(b0p + kt, K, lane);
    v16h b1 = load_b_frag(b1p + kt, K, lane);
    acc[0][0] = wmma_f32(a0, b0, acc[0][0]);
    acc[0][1] = wmma_f32(a0, b1, acc[0][1]);
    acc[1][0] = wmma_f32(a1, b0, acc[1][0]);
    acc[1][1] = wmma_f32(a1, b1, acc[1][1]);
  }
}

// ---------------- elementwise / LN kernels ---------------------------------

__global__ void cvt_f32_to_f16_kernel(const float* __restrict__ in,
                                      _Float16* __restrict__ out, int n) {
  int i = blockIdx.x * 256 + threadIdx.x;
  if (i < n) out[i] = (_Float16)in[i];
}

// One wave32 per 256-element row; 8 rows per 256-thread block.
__global__ void layernorm_f16_kernel(const float* __restrict__ x,
                                     const float* __restrict__ g,
                                     const float* __restrict__ b,
                                     _Float16* __restrict__ out) {
  int wave = threadIdx.x >> 5;
  int lane = threadIdx.x & 31;
  int row = blockIdx.x * 8 + wave;
  const float* xr = x + (size_t)row * DM;
  float v[8];
  float s = 0.f;
#pragma unroll
  for (int i = 0; i < 8; ++i) { v[i] = xr[lane * 8 + i]; s += v[i]; }
#pragma unroll
  for (int off = 16; off > 0; off >>= 1) s += __shfl_xor(s, off, 32);
  float mu = s * (1.0f / DM);
  float q = 0.f;
#pragma unroll
  for (int i = 0; i < 8; ++i) { float d = v[i] - mu; q += d * d; }
#pragma unroll
  for (int off = 16; off > 0; off >>= 1) q += __shfl_xor(q, off, 32);
  float inv = rsqrtf(q * (1.0f / DM) + 1e-5f);
  _Float16* orow = out + (size_t)row * DM;
#pragma unroll
  for (int i = 0; i < 8; ++i) {
    int c = lane * 8 + i;
    orow[c] = (_Float16)((v[i] - mu) * inv * g[c] + b[c]);
  }
}

// ---------------- QKV GEMM: [8192,256]f16 x [768,256]f16^T -----------------
// Epilogue scatters into Q[bh][s][32], K[bh][s][32], Vt[bh][32][s] (f16).

__global__ void gemm_qkv_kernel(const _Float16* __restrict__ yh,
                                const _Float16* __restrict__ wq,
                                const float* __restrict__ bq,
                                _Float16* __restrict__ Qb,
                                _Float16* __restrict__ Kb,
                                _Float16* __restrict__ Vt) {
  int lane = threadIdx.x & 31, w = threadIdx.x >> 5;
  int row0 = blockIdx.x * 128 + (w >> 1) * 32;
  int col0 = blockIdx.y * 64 + (w & 1) * 32;
  v8f acc[2][2] = {};
  gemm_core(yh, wq, DM, row0, col0, lane, acc);
  int mb = (lane >> 4) << 3, nb = lane & 15;
#pragma unroll
  for (int i = 0; i < 2; ++i)
#pragma unroll
    for (int j = 0; j < 2; ++j)
#pragma unroll
      for (int r = 0; r < 8; ++r) {
        int gm = row0 + i * 16 + mb + r;
        int gn = col0 + j * 16 + nb;
        float val = acc[i][j][r] + bq[gn];
        int which = gn >> 8, rem = gn & 255;
        int h = rem >> 5, d = rem & 31;
        int bh = (gm >> 11) * NH + h;
        int s = gm & (SEQ - 1);
        _Float16 hv = (_Float16)val;
        if (which == 0)       Qb[((size_t)bh * SEQ + s) * DH + d] = hv;
        else if (which == 1)  Kb[((size_t)bh * SEQ + s) * DH + d] = hv;
        else                  Vt[((size_t)bh * DH + d) * SEQ + s] = hv;
      }
}

// ---------------- attention: one block per (b,h, 16-query tile) ------------
// dyn LDS: scores 16x2048 f32 (128KB) + ctx accumulator 16x32 f32.

#define ATTN_SMEM_FLOATS (16 * SEQ + 16 * DH)
#define ATTN_SMEM_BYTES  (ATTN_SMEM_FLOATS * 4)

__global__ void attention_kernel(const _Float16* __restrict__ Qb,
                                 const _Float16* __restrict__ Kb,
                                 const _Float16* __restrict__ Vt,
                                 _Float16* __restrict__ ctx) {
  extern __shared__ float sm[];
  float* sc = sm;                   // [16][2048]
  float* ctxacc = sm + 16 * SEQ;    // [16][32]
  __shared__ float red[16][17];
  __shared__ float rowstat[16];

  int tid = threadIdx.x, lane = tid & 31, w = tid >> 5;
  int bh = blockIdx.x >> 7;         // 128 query tiles per (b,h)
  int qt = blockIdx.x & 127;
  int s0 = qt * 16;

  for (int i = tid; i < 16 * DH; i += 256) ctxacc[i] = 0.f;

  // ---- phase 1: scores = (Q K^T) * scale, full 2048-key row in LDS ----
  const _Float16* qbase = Qb + ((size_t)bh * SEQ + s0) * DH;
  v16h qa = load_a_frag(qbase, DH, lane);   // 16 rows x K=32 (=dh)
  int mb = (lane >> 4) << 3, nb = lane & 15;
  const float scale = 0.17677669529663687f; // 1/sqrt(32)
  for (int t = 0; t < 16; ++t) {
    int key0 = w * 256 + t * 16;
    v16h kf = load_b_frag(Kb + ((size_t)bh * SEQ + key0) * DH, DH, lane);
    v8f c = {};
    c = wmma_f32(qa, kf, c);
#pragma unroll
    for (int r = 0; r < 8; ++r)
      sc[(mb + r) * SEQ + key0 + nb] = c[r] * scale;
  }
  __syncthreads();

  // ---- phase 2: softmax over each of the 16 rows (16 threads per row) ----
  int ty = tid >> 4, tx = tid & 15;
  float* rowp = sc + ty * SEQ;
  float lm = -3.0e38f;
  for (int i = 0; i < 128; ++i) lm = fmaxf(lm, rowp[tx + i * 16]);
  red[ty][tx] = lm;
  __syncthreads();
  if (tx == 0) {
    float mm = red[ty][0];
    for (int i = 1; i < 16; ++i) mm = fmaxf(mm, red[ty][i]);
    rowstat[ty] = mm;
  }
  __syncthreads();
  float rm = rowstat[ty];
  float ls = 0.f;
  for (int i = 0; i < 128; ++i) {
    int c0 = tx + i * 16;
    float e = __expf(rowp[c0] - rm);
    rowp[c0] = e;
    ls += e;
  }
  red[ty][tx] = ls;
  __syncthreads();
  if (tx == 0) {
    float ss = 0.f;
    for (int i = 0; i < 16; ++i) ss += red[ty][i];
    rowstat[ty] = 1.f / ss;
  }
  __syncthreads();
  float inv = rowstat[ty];
  for (int i = 0; i < 128; ++i) rowp[tx + i * 16] *= inv;
  __syncthreads();

  // ---- phase 3: ctx = P @ V. wave -> (dh half, key quarter); ds_add ------
  int nhh = w & 1, ks = w >> 1;
  v8f acc = {};
  int kb8 = (lane >> 4) << 3;
  int kb16 = (lane >> 4) << 4;
  int mrow = lane & 15;
  const _Float16* vbase = Vt + ((size_t)bh * DH + nhh * 16 + nb) * SEQ;
  for (int t = 0; t < 16; ++t) {
    int key0 = ks * 512 + t * 32;
    v16h pa;
    const float* prow = sc + mrow * SEQ + key0 + kb8;
#pragma unroll
    for (int e = 0; e < 8; ++e) {
      pa[e]     = (_Float16)prow[e];
      pa[e + 8] = (_Float16)prow[16 + e];
    }
    const _Float16* vp = vbase + key0 + kb8;
    v8h v0 = *(const v8h*)vp;
    v8h v1 = *(const v8h*)(vp + 16);
    v16h vb = __builtin_shufflevector(v0, v1, 0, 1, 2, 3, 4, 5, 6, 7,
                                      8, 9, 10, 11, 12, 13, 14, 15);
    acc = wmma_f32(pa, vb, acc);
  }
#pragma unroll
  for (int r = 0; r < 8; ++r)
    atomicAdd(&ctxacc[(mb + r) * DH + nhh * 16 + nb], acc[r]);
  __syncthreads();

  // ---- writeout: ctx_h[b*2048+s][h*32+d] ----
  int b = bh >> 3, h = bh & 7;
  for (int i = tid; i < 16 * DH; i += 256) {
    int row = i >> 5, col = i & 31;
    size_t gm = (size_t)b * SEQ + s0 + row;
    ctx[gm * DM + h * DH + col] = (_Float16)ctxacc[i];
  }
}

// ---------------- proj / FFN GEMMs -----------------------------------------

__global__ void gemm_proj_kernel(const _Float16* __restrict__ ctx,
                                 const _Float16* __restrict__ wp,
                                 const float* __restrict__ bp,
                                 const float* __restrict__ xin,
                                 float* __restrict__ x1) {
  int lane = threadIdx.x & 31, w = threadIdx.x >> 5;
  int row0 = blockIdx.x * 128 + (w >> 1) * 32;
  int col0 = blockIdx.y * 64 + (w & 1) * 32;
  v8f acc[2][2] = {};
  gemm_core(ctx, wp, DM, row0, col0, lane, acc);
  int mb = (lane >> 4) << 3, nb = lane & 15;
#pragma unroll
  for (int i = 0; i < 2; ++i)
#pragma unroll
    for (int j = 0; j < 2; ++j)
#pragma unroll
      for (int r = 0; r < 8; ++r) {
        size_t gm = row0 + i * 16 + mb + r;
        int gn = col0 + j * 16 + nb;
        x1[gm * DM + gn] = xin[gm * DM + gn] + acc[i][j][r] + bp[gn];
      }
}

__global__ void gemm_ffn1_kernel(const _Float16* __restrict__ y2h,
                                 const _Float16* __restrict__ w1h,
                                 const float* __restrict__ b1,
                                 _Float16* __restrict__ Hh) {
  int lane = threadIdx.x & 31, w = threadIdx.x >> 5;
  int row0 = blockIdx.x * 128 + (w >> 1) * 32;
  int col0 = blockIdx.y * 64 + (w & 1) * 32;
  v8f acc[2][2] = {};
  gemm_core(y2h, w1h, DM, row0, col0, lane, acc);
  int mb = (lane >> 4) << 3, nb = lane & 15;
#pragma unroll
  for (int i = 0; i < 2; ++i)
#pragma unroll
    for (int j = 0; j < 2; ++j)
#pragma unroll
      for (int r = 0; r < 8; ++r) {
        size_t gm = row0 + i * 16 + mb + r;
        int gn = col0 + j * 16 + nb;
        float v = acc[i][j][r] + b1[gn];
        v = v / (1.f + __expf(-v));          // SiLU
        Hh[gm * DFF + gn] = (_Float16)v;
      }
}

__global__ void gemm_ffn2_kernel(const _Float16* __restrict__ Hh,
                                 const _Float16* __restrict__ w2h,
                                 const float* __restrict__ b2,
                                 const float* __restrict__ x1,
                                 float* __restrict__ out) {
  int lane = threadIdx.x & 31, w = threadIdx.x >> 5;
  int row0 = blockIdx.x * 128 + (w >> 1) * 32;
  int col0 = blockIdx.y * 64 + (w & 1) * 32;
  v8f acc[2][2] = {};
  gemm_core(Hh, w2h, DFF, row0, col0, lane, acc);
  int mb = (lane >> 4) << 3, nb = lane & 15;
#pragma unroll
  for (int i = 0; i < 2; ++i)
#pragma unroll
    for (int j = 0; j < 2; ++j)
#pragma unroll
      for (int r = 0; r < 8; ++r) {
        size_t gm = row0 + i * 16 + mb + r;
        int gn = col0 + j * 16 + nb;
        out[gm * DM + gn] = x1[gm * DM + gn] + acc[i][j][r] + b2[gn];
      }
}

// ---------------- host launcher --------------------------------------------

extern "C" void kernel_launch(void* const* d_in, const int* in_sizes, int n_in,
                              void* d_out, int out_size, void* d_ws, size_t ws_size,
                              hipStream_t stream) {
  (void)in_sizes; (void)n_in; (void)out_size; (void)ws_size;

  const float* x      = (const float*)d_in[0];
  const float* ln1_g  = (const float*)d_in[1];
  const float* ln1_b  = (const float*)d_in[2];
  const float* w_qkv  = (const float*)d_in[3];
  const float* b_qkv  = (const float*)d_in[4];
  const float* w_proj = (const float*)d_in[5];
  const float* b_proj = (const float*)d_in[6];
  const float* ln2_g  = (const float*)d_in[7];
  const float* ln2_b  = (const float*)d_in[8];
  const float* w1     = (const float*)d_in[9];
  const float* b1     = (const float*)d_in[10];
  const float* w2     = (const float*)d_in[11];
  const float* b2     = (const float*)d_in[12];
  float* out = (float*)d_out;

  char* ws = (char*)d_ws;
  const size_t MB = 1048576;
  _Float16* yh     = (_Float16*)(ws + 0 * MB);   // 8192x256 f16
  _Float16* Qb     = (_Float16*)(ws + 4 * MB);   // 32x2048x32 f16
  _Float16* Kb     = (_Float16*)(ws + 8 * MB);
  _Float16* Vt     = (_Float16*)(ws + 12 * MB);  // 32x32x2048 f16
  _Float16* ctxh   = (_Float16*)(ws + 16 * MB);  // 8192x256 f16
  float*    X1     = (float*)   (ws + 20 * MB);  // 8192x256 f32
  _Float16* y2h    = (_Float16*)(ws + 28 * MB);  // 8192x256 f16
  _Float16* Hh     = (_Float16*)(ws + 32 * MB);  // 8192x512 f16
  _Float16* wqkvh  = (_Float16*)(ws + 40 * MB);                 // 768x256
  _Float16* wprojh = (_Float16*)(ws + 40 * MB + 393216);        // 256x256
  _Float16* w1h    = (_Float16*)(ws + 40 * MB + 524288);        // 512x256
  _Float16* w2h    = (_Float16*)(ws + 40 * MB + 786432);        // 256x512

  // weight conversion fp32 -> f16
  cvt_f32_to_f16_kernel<<<768, 256, 0, stream>>>(w_qkv, wqkvh, 768 * 256);
  cvt_f32_to_f16_kernel<<<256, 256, 0, stream>>>(w_proj, wprojh, 256 * 256);
  cvt_f32_to_f16_kernel<<<512, 256, 0, stream>>>(w1, w1h, 512 * 256);
  cvt_f32_to_f16_kernel<<<512, 256, 0, stream>>>(w2, w2h, 256 * 512);

  // LN1
  layernorm_f16_kernel<<<MROWS / 8, 256, 0, stream>>>(x, ln1_g, ln1_b, yh);
  // QKV projection
  gemm_qkv_kernel<<<dim3(MROWS / 128, 768 / 64), 256, 0, stream>>>(
      yh, wqkvh, b_qkv, Qb, Kb, Vt);
  // attention
  attention_kernel<<<NB * NH * (SEQ / 16), 256, ATTN_SMEM_BYTES, stream>>>(
      Qb, Kb, Vt, ctxh);
  // output projection + residual
  gemm_proj_kernel<<<dim3(MROWS / 128, DM / 64), 256, 0, stream>>>(
      ctxh, wprojh, b_proj, x, X1);
  // LN2
  layernorm_f16_kernel<<<MROWS / 8, 256, 0, stream>>>(X1, ln2_g, ln2_b, y2h);
  // FFN
  gemm_ffn1_kernel<<<dim3(MROWS / 128, DFF / 64), 256, 0, stream>>>(
      y2h, w1h, b1, Hh);
  gemm_ffn2_kernel<<<dim3(MROWS / 128, DM / 64), 256, 0, stream>>>(
      Hh, w2h, b2, X1, out);
}
